// MultiHeadAttention_84387517432163
// MI455X (gfx1250) — matrix operand order repaired and checked by reading, not compiled, not run on hardware
//
#include <hip/hip_runtime.h>
#include <hip/hip_bf16.h>

// Problem constants (match reference)
#define BB   4
#define SS   2048
#define DD   1024
#define HH   16
#define DKK  64

typedef __attribute__((ext_vector_type(16))) __bf16 v16bf;
typedef __attribute__((ext_vector_type(8)))  __bf16 v8bf;
typedef __attribute__((ext_vector_type(8)))  float  v8f;
typedef __attribute__((ext_vector_type(4)))  float  v4f;

// Native f32 -> bf16 (RNE); lowers to v_cvt_pk_bf16_f32 on gfx1250.
static __device__ __forceinline__ __bf16 f2bf(float f) {
    return (__bf16)f;
}
static __device__ __forceinline__ v16bf cat8(v8bf lo, v8bf hi) {
    return __builtin_shufflevector(lo, hi, 0,1,2,3,4,5,6,7,8,9,10,11,12,13,14,15);
}
static __device__ __forceinline__ v8f wmma_bf16(v16bf a, v16bf b, v8f c) {
    // D = A(16x32) * B(32x16) + C, f32 accumulate
    return __builtin_amdgcn_wmma_f32_16x16x32_bf16(false, a, false, b, (short)0, c, false, false);
}

// Load a 16x32 fp32 A-tile slice for this lane and convert to bf16 A-layout.
// (lane r,h; element e holds k = e + 16h)
static __device__ __forceinline__ v16bf load_a_f32(const float* xrow, int kk, int h) {
    v4f a0 = *(const v4f*)(xrow + kk + 16 * h);
    v4f a1 = *(const v4f*)(xrow + kk + 16 * h + 4);
    v4f a2 = *(const v4f*)(xrow + kk + 16 * h + 8);
    v4f a3 = *(const v4f*)(xrow + kk + 16 * h + 12);
    v16bf a;
    #pragma unroll
    for (int j = 0; j < 4; ++j) {
        a[j]      = f2bf(a0[j]);
        a[4 + j]  = f2bf(a1[j]);
        a[8 + j]  = f2bf(a2[j]);
        a[12 + j] = f2bf(a3[j]);
    }
    return a;
}

// ---------------------------------------------------------------------------
// Kernel 0: convert the four fp32 weight matrices to bf16 (packed wq|wk|wv|wo)
// ---------------------------------------------------------------------------
__global__ __launch_bounds__(256) void wconv_kernel(
    const float* __restrict__ wq, const float* __restrict__ wk,
    const float* __restrict__ wv, const float* __restrict__ wo,
    __bf16* __restrict__ dst)
{
    size_t i = (size_t)blockIdx.x * 256 + threadIdx.x;      // 0 .. 4*D*D-1
    const size_t W = (size_t)DD * DD;                       // 2^20
    const float* src = (i < W) ? wq : (i < 2*W) ? wk : (i < 3*W) ? wv : wo;
    dst[i] = f2bf(src[i & (W - 1)]);
}

// ---------------------------------------------------------------------------
// Kernel 1: QKV projections + bias + RoPE (Q,K) + transpose (V)
//   Per wave: 16(M) x 64(N) tile, K-loop over 1024 in steps of 32.
//   Software-pipelined A tile; B tiles batched ahead of the 4 WMMAs.
//   z = 0:Q  1:K  2:V
// ---------------------------------------------------------------------------
__global__ __launch_bounds__(128) void qkv_kernel(
    const float* __restrict__ xq, const float* __restrict__ xk,
    const float* __restrict__ xv,
    const __bf16* __restrict__ wpack,
    const float* __restrict__ bq, const float* __restrict__ bk,
    const float* __restrict__ bv,
    __bf16* __restrict__ Qb, __bf16* __restrict__ Kb, __bf16* __restrict__ Vt)
{
    const int lane = threadIdx.x;
    const int w    = threadIdx.y;
    const int r    = lane & 15;
    const int h    = lane >> 4;
    const int z    = blockIdx.z;
    const int row0 = blockIdx.x * 16;
    const int n0   = (blockIdx.y * 4 + w) * 64;

    const float*  X    = (z == 0) ? xq : (z == 1) ? xk : xv;
    const __bf16* W    = wpack + (size_t)z * DD * DD;
    const float*  bias = (z == 0) ? bq : (z == 1) ? bk : bv;

    const float* xrow = X + (size_t)(row0 + r) * DD;

    v8f acc[4] = {};
    v16bf a = load_a_f32(xrow, 0, h);
    for (int kk = 0; kk < DD; kk += 32) {
        // Batch all 8 B-tile b128 loads before the WMMA group.
        v16bf bm[4];
        #pragma unroll
        for (int t = 0; t < 4; ++t)
            bm[t] = *(const v16bf*)(W + (size_t)(n0 + 16 * t + r) * DD + kk + 16 * h);
        // Prefetch next A tile into fresh registers (clamped: no OOB).
        const int kn = (kk + 32 < DD) ? kk + 32 : kk;
        v16bf a_next = load_a_f32(xrow, kn, h);
        #pragma unroll
        for (int t = 0; t < 4; ++t)
            acc[t] = wmma_bf16(a, bm[t], acc[t]);
        a = a_next;
    }

    // Epilogue: bias, RoPE (z<2) or transpose-store (z==2)
    #pragma unroll
    for (int t = 0; t < 4; ++t) {
        const int d    = n0 + 16 * t + r;   // output column (C layout: col = lane&15)
        const float bval = bias[d];
        const int head = d >> 6;
        const int dk   = d & 63;
        #pragma unroll
        for (int i = 0; i < 8; ++i) {
            const int R = row0 + i + 8 * h; // C layout: row = reg + 8*half
            const int b = R >> 11;
            const int s = R & (SS - 1);
            float val = acc[t][i] + bval;
            if (z < 2) {
                float other = __shfl_xor(val, 1, 32);   // paired column (d ^ 1)
                float theta = (float)s *
                    __expf(-(float)(dk & 62) * (9.210340371976184f / 64.0f));
                float sn, cs;
                __sincosf(theta, &sn, &cs);
                float out = (dk & 1) ? (other * sn + val * cs)
                                     : (val * cs - other * sn);
                __bf16* dst = (z == 0) ? Qb : Kb;
                dst[((size_t)(b * HH + head) * SS + s) * DKK + dk] = f2bf(out);
            } else {
                Vt[((size_t)(b * HH + head) * DKK + dk) * SS + s] = f2bf(val);
            }
        }
    }
}

// ---------------------------------------------------------------------------
// Kernel 2: flash attention.  One wave per 16-query tile; streams over keys
// in chunks of 32 with online softmax.  8 WMMAs per chunk.  All K/V tiles for
// a chunk are loaded up front so V loads overlap the softmax VALU work.
// ---------------------------------------------------------------------------
__global__ __launch_bounds__(128) void attn_kernel(
    const __bf16* __restrict__ Qb, const __bf16* __restrict__ Kb,
    const __bf16* __restrict__ Vt, __bf16* __restrict__ ctxb)
{
    __shared__ __attribute__((aligned(32))) __bf16 ldsP[4][16][32];

    const int lane = threadIdx.x;
    const int w    = threadIdx.y;
    const int r    = lane & 15;
    const int h    = lane >> 4;
    const int qt   = blockIdx.x * 4 + w;      // global q-tile id
    const int bh   = qt >> 7;                 // = b*H + head  (128 tiles per bh)
    const int q0   = (qt & 127) * 16;
    const int b    = bh >> 4;
    const int head = bh & 15;

    // Q as A operand (rows q0..q0+15, K = dk 0..63 => two 16x32 tiles)
    const __bf16* qrow = Qb + ((size_t)bh * SS + q0 + r) * DKK;
    v16bf aQ0 = cat8(*(const v8bf*)(qrow + 16 * h),      *(const v8bf*)(qrow + 8 + 16 * h));
    v16bf aQ1 = cat8(*(const v8bf*)(qrow + 32 + 16 * h), *(const v8bf*)(qrow + 40 + 16 * h));

    float m_i[8], l_i[8];
    #pragma unroll
    for (int i = 0; i < 8; ++i) { m_i[i] = -1e30f; l_i[i] = 0.0f; }
    v8f acc[4] = {};

    for (int s0 = 0; s0 < SS; s0 += 32) {
        // ---- batch all K and V loads for this 32-key chunk ----
        v16bf bK[4];                           // [sub*2 + dk-half]
        #pragma unroll
        for (int sub = 0; sub < 2; ++sub) {
            const __bf16* krow = Kb + ((size_t)bh * SS + s0 + 16 * sub + r) * DKK;
            bK[2 * sub + 0] = *(const v16bf*)(krow + 16 * h);        // dk 0..31
            bK[2 * sub + 1] = *(const v16bf*)(krow + 32 + 16 * h);   // dk 32..63
        }
        v16bf bV[4];                           // rows of V^T, contiguous in s
        #pragma unroll
        for (int t = 0; t < 4; ++t)
            bV[t] = *(const v16bf*)(Vt + ((size_t)bh * DKK + 16 * t + r) * SS + s0 + 16 * h);

        // ---- scores: two 16x16 tiles (32 keys) ----
        v8f sc[2];
        #pragma unroll
        for (int sub = 0; sub < 2; ++sub) {
            v8f c = {};
            c = wmma_bf16(aQ0, bK[2 * sub + 0], c);
            c = wmma_bf16(aQ1, bK[2 * sub + 1], c);
            sc[sub] = c;
        }

        // ---- online softmax (per-row stats via 16-lane butterflies) ----
        float p0[8], p1[8], corr[8];
        #pragma unroll
        for (int i = 0; i < 8; ++i) {
            float x0 = sc[0][i] * 0.125f;   // 1/sqrt(64)
            float x1 = sc[1][i] * 0.125f;
            float mx = fmaxf(x0, x1);
            #pragma unroll
            for (int md = 1; md < 16; md <<= 1)
                mx = fmaxf(mx, __shfl_xor(mx, md, 32));
            float m_new = fmaxf(m_i[i], mx);
            corr[i] = __expf(m_i[i] - m_new);
            p0[i]   = __expf(x0 - m_new);
            p1[i]   = __expf(x1 - m_new);
            float rs = p0[i] + p1[i];
            #pragma unroll
            for (int md = 1; md < 16; md <<= 1)
                rs += __shfl_xor(rs, md, 32);
            l_i[i] = l_i[i] * corr[i] + rs;
            m_i[i] = m_new;
        }

        // ---- relayout P (C layout -> A layout) through LDS ----
        #pragma unroll
        for (int i = 0; i < 8; ++i) {
            ldsP[w][i + 8 * h][r]      = f2bf(p0[i]);
            ldsP[w][i + 8 * h][16 + r] = f2bf(p1[i]);
        }
        asm volatile("s_wait_dscnt 0" ::: "memory");   // LDS is in-order; wait stores
        v16bf aP = cat8(*(const v8bf*)&ldsP[w][r][16 * h],
                        *(const v8bf*)&ldsP[w][r][8 + 16 * h]);
        asm volatile("" ::: "memory");

        // ---- ctx += P @ V ----
        #pragma unroll
        for (int t = 0; t < 4; ++t) {
            v8f c;
            #pragma unroll
            for (int i = 0; i < 8; ++i) c[i] = acc[t][i] * corr[i];
            acc[t] = wmma_bf16(aP, bV[t], c);
        }
    }

    // ---- normalize and store context (bf16, [B,S,D] row-major) ----
    #pragma unroll
    for (int t = 0; t < 4; ++t) {
        const int dk = 16 * t + r;
        #pragma unroll
        for (int i = 0; i < 8; ++i) {
            const int s = q0 + i + 8 * h;
            float o = acc[t][i] / l_i[i];
            ctxb[(size_t)(b * SS + s) * DD + head * DKK + dk] = f2bf(o);
        }
    }
}

// ---------------------------------------------------------------------------
// Kernel 3: out = ctx @ wo^T + bo + residual   (fp32 into d_out)
// ---------------------------------------------------------------------------
__global__ __launch_bounds__(128) void oproj_kernel(
    const __bf16* __restrict__ ctxb, const __bf16* __restrict__ wob,
    const float* __restrict__ bo, const float* __restrict__ resid,
    float* __restrict__ out)
{
    const int lane = threadIdx.x;
    const int w    = threadIdx.y;
    const int r    = lane & 15;
    const int h    = lane >> 4;
    const int row0 = blockIdx.x * 16;
    const int n0   = (blockIdx.y * 4 + w) * 64;

    const __bf16* arow = ctxb + (size_t)(row0 + r) * DD;

    v8f acc[4] = {};
    v16bf a = cat8(*(const v8bf*)(arow + 16 * h),
                   *(const v8bf*)(arow + 8 + 16 * h));
    for (int kk = 0; kk < DD; kk += 32) {
        v16bf bm[4];
        #pragma unroll
        for (int t = 0; t < 4; ++t)
            bm[t] = *(const v16bf*)(wob + (size_t)(n0 + 16 * t + r) * DD + kk + 16 * h);
        const int kn = (kk + 32 < DD) ? kk + 32 : kk;
        v16bf a_next = cat8(*(const v8bf*)(arow + kn + 16 * h),
                            *(const v8bf*)(arow + kn + 8 + 16 * h));
        #pragma unroll
        for (int t = 0; t < 4; ++t)
            acc[t] = wmma_bf16(a, bm[t], acc[t]);
        a = a_next;
    }
    #pragma unroll
    for (int t = 0; t < 4; ++t) {
        const int d = n0 + 16 * t + r;
        const float bval = bo[d];
        #pragma unroll
        for (int i = 0; i < 8; ++i) {
            const size_t R = row0 + i + 8 * h;
            out[R * DD + d] = acc[t][i] + bval + resid[R * DD + d];
        }
    }
}

// ---------------------------------------------------------------------------
// Kernel 4: in-place LayerNorm over the last dim (1024), one block per row
// ---------------------------------------------------------------------------
__global__ __launch_bounds__(256) void ln_kernel(
    float* __restrict__ x,
    const float* __restrict__ gamma, const float* __restrict__ beta)
{
    __shared__ float rs[8], rs2[8];
    const int tid = threadIdx.x;
    float* row = x + (size_t)blockIdx.x * DD;

    float v[4];
    float s = 0.0f, s2 = 0.0f;
    #pragma unroll
    for (int j = 0; j < 4; ++j) {
        v[j] = row[tid + 256 * j];
        s  += v[j];
        s2 += v[j] * v[j];
    }
    #pragma unroll
    for (int m = 1; m < 32; m <<= 1) {
        s  += __shfl_xor(s,  m, 32);
        s2 += __shfl_xor(s2, m, 32);
    }
    if ((tid & 31) == 0) { rs[tid >> 5] = s; rs2[tid >> 5] = s2; }
    __syncthreads();
    float ts = 0.0f, ts2 = 0.0f;
    #pragma unroll
    for (int k = 0; k < 8; ++k) { ts += rs[k]; ts2 += rs2[k]; }
    const float mu   = ts * (1.0f / DD);
    const float var  = ts2 * (1.0f / DD) - mu * mu;
    const float rstd = rsqrtf(var + 1e-5f);
    #pragma unroll
    for (int j = 0; j < 4; ++j) {
        const int d = tid + 256 * j;
        row[d] = (v[j] - mu) * rstd * gamma[d] + beta[d];
    }
}

// ---------------------------------------------------------------------------
extern "C" void kernel_launch(void* const* d_in, const int* in_sizes, int n_in,
                              void* d_out, int out_size, void* d_ws, size_t ws_size,
                              hipStream_t stream)
{
    (void)in_sizes; (void)n_in; (void)out_size; (void)ws_size;

    const float* query = (const float*)d_in[0];
    const float* key_  = (const float*)d_in[1];
    const float* value = (const float*)d_in[2];
    const float* wq    = (const float*)d_in[3];
    const float* bq    = (const float*)d_in[4];
    const float* wk    = (const float*)d_in[5];
    const float* bk    = (const float*)d_in[6];
    const float* wv    = (const float*)d_in[7];
    const float* bv    = (const float*)d_in[8];
    const float* wo    = (const float*)d_in[9];
    const float* bo    = (const float*)d_in[10];
    const float* gamma = (const float*)d_in[11];
    const float* beta  = (const float*)d_in[12];
    float* out = (float*)d_out;

    const size_t WSZ = (size_t)DD * DD;             // 1M elems per weight
    const size_t QSZ = (size_t)BB * HH * SS * DKK;  // 8M elems per tensor

    __bf16* wpack = (__bf16*)d_ws;          // wq|wk|wv|wo  (4*WSZ bf16)
    __bf16* Qb    = wpack + 4 * WSZ;        // [B,H,S,DK]
    __bf16* Kb    = Qb + QSZ;               // [B,H,S,DK]
    __bf16* Vt    = Kb + QSZ;               // [B,H,DK,S]
    __bf16* ctxb  = Vt + QSZ;               // [B,S,D]
    // total ws: (4*1M + 4*8M) * 2B = 72 MB

    wconv_kernel<<<(4 * WSZ) / 256, 256, 0, stream>>>(wq, wk, wv, wo, wpack);

    qkv_kernel<<<dim3(BB * SS / 16, DD / 256, 3), dim3(32, 4), 0, stream>>>(
        query, key_, value, wpack, bq, bk, bv, Qb, Kb, Vt);

    attn_kernel<<<dim3(BB * HH * (SS / 16) / 4), dim3(32, 4), 0, stream>>>(
        Qb, Kb, Vt, ctxb);

    oproj_kernel<<<dim3(BB * SS / 16, DD / 256), dim3(32, 4), 0, stream>>>(
        ctxb, wpack + 3 * WSZ, bo, query, out);

    ln_kernel<<<BB * SS, 256, 0, stream>>>(out, gamma, beta);
}
